// Attention_42657615184259
// MI455X (gfx1250) — matrix operand order repaired and checked by reading, not compiled, not run on hardware
//
#include <hip/hip_runtime.h>
#include <hip/hip_bf16.h>

// MHA forward: B=8, N=2048, DIM=384, HEADS=6, DK=64, SCALE=1/8.
// Pipeline: qkv_gemm (f32->f16, WMMA) -> flash attention (WMMA + online softmax) -> proj_gemm.

typedef _Float16 h16;
typedef __attribute__((ext_vector_type(8)))  _Float16 v8h;
typedef __attribute__((ext_vector_type(16))) _Float16 v16h;
typedef __attribute__((ext_vector_type(8)))  float    v8f;

__device__ __forceinline__ v8f wmma_f16(v16h a, v16h b, v8f c) {
  return __builtin_amdgcn_wmma_f32_16x16x32_f16(false, a, false, b, (short)0, c,
                                                false, false);
}

// Build a 16-half A/B fragment from two 16-byte LDS/global chunks.
__device__ __forceinline__ v16h frag16(const h16* lo, const h16* hi) {
  v8h a = *(const v8h*)lo;
  v8h b = *(const v8h*)hi;
  v16h r;
#pragma unroll
  for (int i = 0; i < 8; ++i) { r[i] = a[i]; r[i + 8] = b[i]; }
  return r;
}

__device__ __forceinline__ v8h cvt8(const float* p) {
  float4 f0 = *(const float4*)p;
  float4 f1 = *(const float4*)(p + 4);
  v8h t;
  t[0] = (h16)f0.x; t[1] = (h16)f0.y; t[2] = (h16)f0.z; t[3] = (h16)f0.w;
  t[4] = (h16)f1.x; t[5] = (h16)f1.y; t[6] = (h16)f1.z; t[7] = (h16)f1.w;
  return t;
}

// Async global->LDS 16-byte copy (no VGPR round trip).  LDS operand is the low
// 32 bits of the generic pointer (LDS aperture maps addr[31:0] to DS address).
// Tracked with ASYNCcnt; pair with async_wait() before the workgroup barrier.
__device__ __forceinline__ void async_copy16(const h16* g, h16* l) {
  unsigned lds = (unsigned)(unsigned long long)l;
  asm volatile("global_load_async_to_lds_b128 %0, %1, off"
               :: "v"(lds), "v"(g) : "memory");
}
__device__ __forceinline__ void async_wait() {
  asm volatile("s_wait_asynccnt 0x0" ::: "memory");
}

// ---------------------------------------------------------------------------
// Kernel 1: qkv = x @ qkv_w^T + qkv_b  -> head-split f16 Q (pre-scaled), K, V
// M = B*N = 16384, Nout = 1152, K = 384.  Block: 64x64 tile, 8 waves.
// ---------------------------------------------------------------------------
__global__ __launch_bounds__(256) void qkv_gemm(const float* __restrict__ x,
                                                const float* __restrict__ w,
                                                const float* __restrict__ bias,
                                                h16* __restrict__ Qo,
                                                h16* __restrict__ Ko,
                                                h16* __restrict__ Vo) {
  __shared__ __align__(16) h16 As[64][40];  // [m][k], stride 40 halves = 80B
  __shared__ __align__(16) h16 Bs[64][40];  // [n][k]
  const int tid  = threadIdx.x;
  const int lane = tid & 31, wv = tid >> 5;
  const int mblk = blockIdx.x, nblk = blockIdx.y;
  const int mr = lane & 15, ksel = lane >> 4;
  const int trow = (wv >> 2) * 16;  // wave's tile0 row base; tile1 = +32
  const int tcol = (wv & 3) * 16;
  v8f c0 = {}, c1 = {};

  const int lr = tid >> 2, ls = (tid & 3) * 8;  // cooperative-load coords
  const float* xrow = x + (size_t)(mblk * 64 + lr) * 384 + ls;
  const float* wrow = w + (size_t)(nblk * 64 + lr) * 384 + ls;

  for (int k0 = 0; k0 < 384; k0 += 32) {
    *(v8h*)&As[lr][ls] = cvt8(xrow + k0);
    *(v8h*)&Bs[lr][ls] = cvt8(wrow + k0);
    __builtin_prefetch(xrow + k0 + 32, 0, 0);
    __builtin_prefetch(wrow + k0 + 32, 0, 0);
    __syncthreads();
    // A frag: lane holds row (trow+mr); halves 0-7 -> k=8*ksel.., 8-15 -> k=16+8*ksel..
    const h16* ap0 = &As[trow + mr][8 * ksel];
    const h16* ap1 = &As[trow + 32 + mr][8 * ksel];
    // B frag: lane holds col (tcol+mr); halves 0-7 -> k=8*ksel.., 8-15 -> k=16+8*ksel..
    const h16* bp = &Bs[tcol + mr][8 * ksel];
    v16h bf = frag16(bp, bp + 16);
    c0 = wmma_f16(frag16(ap0, ap0 + 16), bf, c0);
    c1 = wmma_f16(frag16(ap1, ap1 + 16), bf, c1);
    __syncthreads();
  }

  // Epilogue: C layout -> lane col = tcol+mr, VGPR r -> row = r + 8*ksel.
  const int col = tcol + mr;
  const int gn  = nblk * 64 + col;
  const float bv = bias[gn];
  const int t3 = nblk / 6, hh = nblk % 6;  // 0:Q 1:K 2:V ; head index
  h16* dst = (t3 == 0) ? Qo : (t3 == 1) ? Ko : Vo;
  const float scl = (t3 == 0) ? 0.125f : 1.0f;  // fold SCALE into Q
#pragma unroll
  for (int r = 0; r < 8; ++r) {
    int gm = mblk * 64 + trow + r + 8 * ksel;
    int bb = gm >> 11, nn = gm & 2047;
    dst[((size_t)(bb * 6 + hh) * 2048 + nn) * 64 + col] = (h16)((c0[r] + bv) * scl);
    gm += 32; bb = gm >> 11; nn = gm & 2047;
    dst[((size_t)(bb * 6 + hh) * 2048 + nn) * 64 + col] = (h16)((c1[r] + bv) * scl);
  }
}

// ---------------------------------------------------------------------------
// Kernel 2: flash attention.  One wave per 16-query tile; 8 waves per block
// share one (b,h) and stream K/V in 32-key chunks through LDS.  K chunks are
// staged with async global->LDS copies (ASYNCcnt path).
// ---------------------------------------------------------------------------
__global__ __launch_bounds__(256) void attn_kernel(const h16* __restrict__ Q,
                                                   const h16* __restrict__ K,
                                                   const h16* __restrict__ V,
                                                   h16* __restrict__ AO) {
  __shared__ __align__(16) h16 Ks[32][72];      // [key][dk]  for Q@K^T B-frags
  __shared__ __align__(16) h16 Vs[64][40];      // [dk][key]  for P@V  B-frags
  __shared__ __align__(16) h16 Ps[8][16][40];   // per-wave P scratch [row][key]
  const int tid  = threadIdx.x;
  const int lane = tid & 31, wv = tid >> 5;
  const int mr = lane & 15, ksel = lane >> 4;
  const int bh = blockIdx.x >> 4, qblk = blockIdx.x & 15;
  const int q0 = qblk * 128 + wv * 16;
  const h16* Qb = Q + (size_t)bh * 2048 * 64;
  const h16* Kb = K + (size_t)bh * 2048 * 64;
  const h16* Vb = V + (size_t)bh * 2048 * 64;

  // Q A-fragments (16 queries x 64 dk -> two 16x32 frags), direct b128 loads.
  const h16* qrow = Qb + (size_t)(q0 + mr) * 64 + 16 * ksel;
  v16h qa0 = frag16(qrow, qrow + 8);         // dk 0..31
  v16h qa1 = frag16(qrow + 32, qrow + 40);   // dk 32..63

  v8f o[4] = {};        // 16 x 64 output accumulator (C layout, 4 dk tiles)
  float mrow[8], lrow[8];
#pragma unroll
  for (int r = 0; r < 8; ++r) { mrow[r] = -1e30f; lrow[r] = 0.0f; }

  const int kr = tid >> 3, kseg = (tid & 7) * 8;  // cooperative-load coords
  for (int ch = 0; ch < 64; ++ch) {
    const int kc0 = ch * 32;
    // Stage K chunk via async copy (no VGPR round trip), V chunk transposed.
    async_copy16(Kb + (size_t)(kc0 + kr) * 64 + kseg, &Ks[kr][kseg]);
    v8h vv = *(const v8h*)(Vb + (size_t)(kc0 + kr) * 64 + kseg);
#pragma unroll
    for (int i = 0; i < 8; ++i) Vs[kseg + i][kr] = vv[i];
    async_wait();
    __syncthreads();

    // S = Q @ K^T for keys [kc0, kc0+32): two 16x16 tiles, 2 WMMAs each.
    v8f s0 = {}, s1 = {};
    {
      const h16* p = &Ks[mr][16 * ksel];
      s0 = wmma_f16(qa0, frag16(p, p + 8), s0);
      const h16* p2 = &Ks[mr][32 + 16 * ksel];
      s0 = wmma_f16(qa1, frag16(p2, p2 + 8), s0);
      const h16* q1 = &Ks[16 + mr][16 * ksel];
      s1 = wmma_f16(qa0, frag16(q1, q1 + 8), s1);
      const h16* q2 = &Ks[16 + mr][32 + 16 * ksel];
      s1 = wmma_f16(qa1, frag16(q2, q2 + 8), s1);
    }

    // Online softmax. Row r of this lane is query (r + 8*ksel); row stats are
    // reduced across the 16-lane half via shfl_xor (1,2,4,8 stay in-half).
    float alpha[8];
#pragma unroll
    for (int r = 0; r < 8; ++r) {
      float mc = fmaxf(s0[r], s1[r]);
      mc = fmaxf(mc, __shfl_xor(mc, 1, 32));
      mc = fmaxf(mc, __shfl_xor(mc, 2, 32));
      mc = fmaxf(mc, __shfl_xor(mc, 4, 32));
      mc = fmaxf(mc, __shfl_xor(mc, 8, 32));
      float mn = fmaxf(mrow[r], mc);
      alpha[r] = __expf(mrow[r] - mn);
      float p0 = __expf(s0[r] - mn);
      float p1 = __expf(s1[r] - mn);
      float rs = p0 + p1;
      rs += __shfl_xor(rs, 1, 32);
      rs += __shfl_xor(rs, 2, 32);
      rs += __shfl_xor(rs, 4, 32);
      rs += __shfl_xor(rs, 8, 32);
      lrow[r] = lrow[r] * alpha[r] + rs;
      mrow[r] = mn;
      // C layout -> Ps[row][key]: this lane owns column mr of both key tiles.
      const int prow = r + 8 * ksel;
      Ps[wv][prow][mr]      = (h16)p0;
      Ps[wv][prow][16 + mr] = (h16)p1;
    }
#pragma unroll
    for (int r = 0; r < 8; ++r) {
      o[0][r] *= alpha[r]; o[1][r] *= alpha[r];
      o[2][r] *= alpha[r]; o[3][r] *= alpha[r];
    }

    // Re-read P as an A fragment (16 queries x 32 keys).
    const h16* pp = &Ps[wv][mr][16 * ksel];
    v16h pa = frag16(pp, pp + 8);
    // O += P @ V  (V chunk is 32 keys x 64 dk -> 4 B frags from Vs[dk][key]).
#pragma unroll
    for (int j = 0; j < 4; ++j) {
      const h16* vp = &Vs[16 * j + mr][16 * ksel];
      o[j] = wmma_f16(pa, frag16(vp, vp + 8), o[j]);
    }
    __syncthreads();
  }

  // Normalize and write AO[b][n][h*64 + dk] as f16.
  const int bq = bh / 6, hh = bh % 6;
#pragma unroll
  for (int r = 0; r < 8; ++r) {
    const float inv = 1.0f / lrow[r];
    const int qg = q0 + r + 8 * ksel;
    const size_t base = ((size_t)bq * 2048 + qg) * 384 + hh * 64 + mr;
    AO[base]      = (h16)(o[0][r] * inv);
    AO[base + 16] = (h16)(o[1][r] * inv);
    AO[base + 32] = (h16)(o[2][r] * inv);
    AO[base + 48] = (h16)(o[3][r] * inv);
  }
}

// ---------------------------------------------------------------------------
// Kernel 3: out = AO @ proj_w^T + proj_b (f32 output).  M=16384,N=384,K=384.
// A tile (already f16) is staged with async global->LDS copies.
// ---------------------------------------------------------------------------
__global__ __launch_bounds__(256) void proj_gemm(const h16* __restrict__ a,
                                                 const float* __restrict__ w,
                                                 const float* __restrict__ bias,
                                                 float* __restrict__ out) {
  __shared__ __align__(16) h16 As[64][40];
  __shared__ __align__(16) h16 Bs[64][40];
  const int tid  = threadIdx.x;
  const int lane = tid & 31, wv = tid >> 5;
  const int mblk = blockIdx.x, nblk = blockIdx.y;
  const int mr = lane & 15, ksel = lane >> 4;
  const int trow = (wv >> 2) * 16;
  const int tcol = (wv & 3) * 16;
  v8f c0 = {}, c1 = {};

  const int lr = tid >> 2, ls = (tid & 3) * 8;
  const h16*   arow = a + (size_t)(mblk * 64 + lr) * 384 + ls;
  const float* wrow = w + (size_t)(nblk * 64 + lr) * 384 + ls;

  for (int k0 = 0; k0 < 384; k0 += 32) {
    async_copy16(arow + k0, &As[lr][ls]);
    *(v8h*)&Bs[lr][ls] = cvt8(wrow + k0);
    __builtin_prefetch(wrow + k0 + 32, 0, 0);
    async_wait();
    __syncthreads();
    const h16* ap0 = &As[trow + mr][8 * ksel];
    const h16* ap1 = &As[trow + 32 + mr][8 * ksel];
    const h16* bp = &Bs[tcol + mr][8 * ksel];
    v16h bf = frag16(bp, bp + 16);
    c0 = wmma_f16(frag16(ap0, ap0 + 16), bf, c0);
    c1 = wmma_f16(frag16(ap1, ap1 + 16), bf, c1);
    __syncthreads();
  }

  const int gn = nblk * 64 + tcol + mr;
  const float bv = bias[gn];
#pragma unroll
  for (int r = 0; r < 8; ++r) {
    int gm = mblk * 64 + trow + r + 8 * ksel;
    out[(size_t)gm * 384 + gn] = c0[r] + bv;
    gm += 32;
    out[(size_t)gm * 384 + gn] = c1[r] + bv;
  }
}

// ---------------------------------------------------------------------------
extern "C" void kernel_launch(void* const* d_in, const int* in_sizes, int n_in,
                              void* d_out, int out_size, void* d_ws, size_t ws_size,
                              hipStream_t stream) {
  (void)in_sizes; (void)n_in; (void)out_size; (void)ws_size;
  const float* x      = (const float*)d_in[0];
  const float* qkv_w  = (const float*)d_in[1];
  const float* qkv_b  = (const float*)d_in[2];
  const float* proj_w = (const float*)d_in[3];
  const float* proj_b = (const float*)d_in[4];
  float* out = (float*)d_out;

  // Workspace: f16 Q | K | V | AO, each B*H*N*DK = 6,291,456 elements.
  const size_t S1 = (size_t)8 * 6 * 2048 * 64;
  char* ws = (char*)d_ws;
  h16* Qw  = (h16*)(ws);
  h16* Kw  = (h16*)(ws + 2 * S1);
  h16* Vw  = (h16*)(ws + 4 * S1);
  h16* AOw = (h16*)(ws + 6 * S1);

  qkv_gemm <<<dim3(256, 18), 256, 0, stream>>>(x, qkv_w, qkv_b, Qw, Kw, Vw);
  attn_kernel<<<dim3(768),    256, 0, stream>>>(Qw, Kw, Vw, AOw);
  proj_gemm<<<dim3(256, 6),  256, 0, stream>>>(AOw, proj_w, proj_b, out);
}
